// OuterMean_824633721468
// MI455X (gfx1250) — compile-verified, matrix operand order checked by reading, not run on hardware
//
#include <hip/hip_runtime.h>
#include <hip/hip_bf16.h>

typedef __attribute__((ext_vector_type(16))) __bf16 v16bf;
typedef __attribute__((ext_vector_type(8)))  __bf16 v8bf;
typedef __attribute__((ext_vector_type(8)))  float  v8f;
typedef __attribute__((ext_vector_type(2)))  float  v2f;
typedef unsigned int v4u __attribute__((ext_vector_type(4)));
typedef int          v8i __attribute__((ext_vector_type(8)));
typedef int          v4i __attribute__((ext_vector_type(4)));

#define B_  2
#define M_  16
#define I_  512
#define D_  128
#define H_  128
#define TI  32          // I_/16
#define EPS 1e-5f

// workspace layout (bytes)
#define WLW_OFF 0u                     // 4*8*32*16 bf16 = 32 KB  (Wl, B-operand layout)
#define WRW_OFF 32768u                 // 32 KB                    (Wr)
#define WOW_OFF 65536u                 // 32 KB                    (Wo)
#define LW_OFF  98304u                 // [b][it][h][il][m] f32 = 8 MB (masked left proj)
#define RW_OFF  (98304u + 8388608u)    // same, right proj = 8 MB
// total ~16.1 MB of d_ws

#if __has_builtin(__builtin_amdgcn_tensor_load_to_lds)
#define USE_TDM 1
#else
#define USE_TDM 0
#endif

// ---------------------------------------------------------------------------
// Kernel 0: repack 128x128 f32 weights (contraction dim = row) into the bf16
// WMMA B-operand (32x16 k-major) per-lane layout:
//   element idx 0..15 of lane L = k in {0..7,16..23} (+8 if L>=16), n = nt*16+(L&15)
// ---------------------------------------------------------------------------
__global__ __launch_bounds__(256) void prep_weights_kernel(
    const float* __restrict__ Wl, const float* __restrict__ Wr,
    const float* __restrict__ Wo,
    __bf16* __restrict__ wlw, __bf16* __restrict__ wrw, __bf16* __restrict__ wow) {
  for (int e = threadIdx.x; e < 4 * 8 * 32 * 16; e += 256) {
    int idx  = e & 15;
    int lane = (e >> 4) & 31;
    int nt   = (e >> 9) & 7;
    int kt   = e >> 12;
    int k = kt * 32 + (idx < 8 ? idx : idx + 8) + ((lane >> 4) << 3);
    int n = nt * 16 + (lane & 15);
    wlw[e] = (__bf16)Wl[k * 128 + n];
    wrw[e] = (__bf16)Wr[k * 128 + n];
    wow[e] = (__bf16)Wo[k * 128 + n];
  }
}

// ---------------------------------------------------------------------------
// Kernel 1: left/right projections. One wave per (b, m, i-tile).
//   L = mask * (x @ Wl + bl), R = mask * (x @ Wr + br)
// Output layout Lw/Rw: [b][it][h][il(16)][m(16)] f32 so that stage-2a f32 WMMA
// A/B operands are contiguous v2f per lane.
// ---------------------------------------------------------------------------
__global__ __launch_bounds__(32) void proj_kernel(
    const float* __restrict__ x, const int* __restrict__ mask,
    const __bf16* __restrict__ wlw, const float* __restrict__ bl,
    const __bf16* __restrict__ wrw, const float* __restrict__ br,
    float* __restrict__ Lw, float* __restrict__ Rw) {
  int bid = blockIdx.x;
  int it  = bid % TI;
  int m   = (bid / TI) % M_;
  int b   = bid / (TI * M_);
  int lane = threadIdx.x;
  int hi   = lane >> 4;           // lane half selects k-group
  int i0   = it * 16;
  int row  = i0 + (lane & 15);    // A-matrix row = i

  // Load x tile as bf16 A-operand (16 x 32 per k-step), 4 k-steps cover D=128.
  v16bf xa[4];
  const float* xr = x + (((size_t)(b * M_ + m) * I_) + row) * D_;
  #pragma unroll
  for (int kt = 0; kt < 4; ++kt) {
    int dbase = kt * 32 + hi * 8;
    #pragma unroll
    for (int e = 0; e < 8; ++e) {
      xa[kt][e]     = (__bf16)xr[dbase + e];
      xa[kt][8 + e] = (__bf16)xr[dbase + 16 + e];
    }
  }
  // mask for the 8 C-rows this lane owns (C: VGPR r -> row r (+8 for lanes>=16))
  float mrow[8];
  #pragma unroll
  for (int r = 0; r < 8; ++r) {
    int il = r + hi * 8;
    mrow[r] = mask[(b * M_ + m) * I_ + i0 + il] ? 1.0f : 0.0f;
  }

  #pragma unroll
  for (int ht = 0; ht < 8; ++ht) {      // 8 h-tiles of 16
    v8f accL = {}; v8f accR = {};
    #pragma unroll
    for (int kt = 0; kt < 4; ++kt) {
      v16bf wl = *(const v16bf*)(wlw + ((kt * 8 + ht) * 32 + lane) * 16);
      v16bf wr = *(const v16bf*)(wrw + ((kt * 8 + ht) * 32 + lane) * 16);
      accL = __builtin_amdgcn_wmma_f32_16x16x32_bf16(false, xa[kt], false, wl,
                                                     (short)0, accL, false, false);
      accR = __builtin_amdgcn_wmma_f32_16x16x32_bf16(false, xa[kt], false, wr,
                                                     (short)0, accR, false, false);
    }
    int h = ht * 16 + (lane & 15);      // C-col = h
    float blv = bl[h], brv = br[h];
    #pragma unroll
    for (int r = 0; r < 8; ++r) {
      int il = r + hi * 8;
      int o  = (((b * TI + it) * H_ + h) * 16 + il) * 16 + m;
      Lw[o] = (accL[r] + blv) * mrow[r];
      Rw[o] = (accR[r] + brv) * mrow[r];
    }
  }
}

// ---------------------------------------------------------------------------
// Kernel 2 (fused): per (b, i-tile, j-tile), 8 waves.
//  TDM: wave 0 DMAs the staged bf16 Wo (32 KB) into LDS (overlapped with 2a),
//       completed with s_wait_tensorcnt before the block barrier.
//  stage 2a: outer[i,j,h] = (1/(M*(pc[i,j]+eps))) * sum_m L[m,i,h]*R[m,j,h]
//            via v_wmma_f32_16x16x4_f32 (K=m, 4 steps), 16 h per wave.
//            Packed bf16 into LDS in GEMM2 A-operand layout. Each row's h-span
//            is rotated by (row&15)*8 elements (16B units) so b16 writer lanes
//            stride 4 banks apart instead of all hitting bank 0; rotation is a
//            multiple of 8 elements so ds_load_b128 reads stay contiguous.
//  stage 2b: out[row, d] = outer_row @ Wo + bo via v_wmma_f32_16x16x32_bf16,
//            B-operand from the TDM-loaded LDS copy of Wo.
// ---------------------------------------------------------------------------
__global__ __launch_bounds__(256) void outer_out_kernel(
    const float* __restrict__ Lw, const float* __restrict__ Rw,
    const __bf16* __restrict__ wow, const int* __restrict__ mask,
    const float* __restrict__ bo, float* __restrict__ out) {
  __shared__ __align__(16) __bf16 ldsA[256 * 128];   // 64 KB: outer tile (bf16)
  __shared__ __align__(16) __bf16 ldsB[4 * 8 * 32 * 16]; // 32 KB: Wo B-operands

  int bid = blockIdx.x;
  int jt  = bid % TI;
  int it  = (bid / TI) % TI;
  int b   = bid / (TI * TI);
  int t    = threadIdx.x;
  int lane = t & 31;
  int wave = t >> 5;
  int hi   = lane >> 4;
  int loc  = lane & 15;           // j (2a cols) / lds row lane / d lane (2b)
  int i0 = it * 16, j0 = jt * 16;

#if USE_TDM
  if (wave == 0) {
    // Tensor DMA descriptor (D#): 1-D tile of 8192 dwords global->LDS.
    __attribute__((address_space(3))) __bf16* lp =
        (__attribute__((address_space(3))) __bf16*)&ldsB[0];
    unsigned ldsOff = (unsigned)(unsigned long long)lp;
    unsigned long long ga = (unsigned long long)wow;
    v4u g0;
    g0[0] = 1u;                                            // count=1, user mode
    g0[1] = ldsOff;                                        // lds_addr
    g0[2] = (unsigned)(ga & 0xffffffffull);                // global_addr[95:64]
    g0[3] = (unsigned)((ga >> 32) & 0x01ffffffull)         // global_addr[120:96]
            | (2u << 30);                                  // type = 2 ("image")
    v8i g1;
    g1[0] = (int)(2u << 16);        // workgroup_mask=0, data_size=2 (4 B)
    g1[1] = (int)(8192u << 16);     // tensor_dim0 = 8192 dwords (low 16)
    g1[2] = (int)(1u << 16);        // tensor_dim0 hi=0, tensor_dim1 = 1
    g1[3] = (int)(8192u << 16);     // tensor_dim1 hi=0, tile_dim0 = 8192
    g1[4] = 0;                      // tile_dim1 = tile_dim2 = 0 (unused)
    g1[5] = (int)8192u;             // tensor_dim0_stride (low 32)
    g1[6] = 0;
    g1[7] = 0;
    v4i gz = {0, 0, 0, 0};
#if __clang_major__ >= 23
    v8i gz8 = (v8i)0;
    __builtin_amdgcn_tensor_load_to_lds(g0, g1, gz, gz, gz8, 0);
#else
    __builtin_amdgcn_tensor_load_to_lds(g0, g1, gz, gz, 0);
#endif
  }
#else
  // Fallback: cooperative copy of staged Wo into LDS.
  for (int e = t; e < (4 * 8 * 32 * 16) / 8; e += 256)
    ((v8bf*)ldsB)[e] = ((const v8bf*)wow)[e];
#endif

  // Per-thread reciprocal pair-count for the 8 (i,j) cells this lane's C regs hold.
  float pcinv[8];
  {
    int ib = b * M_ * I_;
    int mj[M_];
    #pragma unroll
    for (int mm = 0; mm < M_; ++mm) mj[mm] = mask[ib + mm * I_ + j0 + loc];
    #pragma unroll
    for (int r = 0; r < 8; ++r) {
      int il = r + hi * 8;
      float s = 0.0f;
      #pragma unroll
      for (int mm = 0; mm < M_; ++mm)
        s += (mask[ib + mm * I_ + i0 + il] && mj[mm]) ? 1.0f : 0.0f;
      pcinv[r] = 1.0f / ((float)M_ * (s + EPS));
    }
  }

  // ---- stage 2a: m-contraction in f32 WMMA, 16 h values per wave ----
  const float* Lb = Lw + (size_t)((b * TI + it) * H_) * 256;
  const float* Rb = Rw + (size_t)((b * TI + jt) * H_) * 256;
  #pragma unroll 1
  for (int hh = 0; hh < 16; ++hh) {
    int h = wave * 16 + hh;
    const float* lrow = Lb + (h * 16 + loc) * 16;   // A row = i, 16 m values
    const float* rrow = Rb + (h * 16 + loc) * 16;   // B col = j, 16 m values
    if (hh + 1 < 16) {                               // hint: global_prefetch
      __builtin_prefetch(Lb + ((wave * 16 + hh + 1) * 16 + loc) * 16, 0, 1);
      __builtin_prefetch(Rb + ((wave * 16 + hh + 1) * 16 + loc) * 16, 0, 1);
    }
    v8f c = {};
    #pragma unroll
    for (int ks = 0; ks < 4; ++ks) {     // K = m, 4 per step (f32 WMMA)
      int moff = 4 * ks + hi * 2;        // lanes>=16 carry K=+2,+3
      v2f a  = *(const v2f*)(lrow + moff);
      v2f bm = *(const v2f*)(rrow + moff);
      c = __builtin_amdgcn_wmma_f32_16x16x4_f32(false, a, false, bm,
                                                (short)0, c, false, false);
    }
    #pragma unroll
    for (int r = 0; r < 8; ++r) {
      int il   = r + hi * 8;
      int rowA = il * 16 + loc;          // GEMM2 row = li*16 + lj
      int hs   = (h + ((rowA & 15) << 3)) & 127;     // bank-spread rotation
      ldsA[rowA * 128 + hs] = (__bf16)(c[r] * pcinv[r]);
    }
  }
#if USE_TDM
  if (wave == 0) __builtin_amdgcn_s_wait_tensorcnt(0);
#endif
  __syncthreads();

  // ---- stage 2b: (256 x 128) @ Wo(128 x 128) in bf16 WMMA ----
  #pragma unroll 1
  for (int t8 = 0; t8 < 16; ++t8) {
    int tile = wave * 16 + t8;           // 128 output tiles over 8 waves
    int rt = tile >> 3;                  // row-tile (= li)
    int dt = tile & 7;                   // d-tile
    v8f c = {};
    #pragma unroll
    for (int kt = 0; kt < 4; ++kt) {
      int row  = rt * 16 + loc;
      int rowe = row * 128;              // 256 B row stride -> 16 B aligned
      int rot  = (row & 15) << 3;        // matches writer rotation
      int h0   = kt * 32 + hi * 8;       // lane k-groups {0..7,16..23}/{8..15,24..31}
      v8bf g0 = *(const v8bf*)(ldsA + rowe + ((h0 + rot) & 127));
      v8bf g1 = *(const v8bf*)(ldsA + rowe + ((h0 + 16 + rot) & 127));
      v16bf a;
      #pragma unroll
      for (int e = 0; e < 8; ++e) { a[e] = g0[e]; a[8 + e] = g1[e]; }
      v16bf bm = *(const v16bf*)(ldsB + ((kt * 8 + dt) * 32 + lane) * 16);
      c = __builtin_amdgcn_wmma_f32_16x16x32_bf16(false, a, false, bm,
                                                  (short)0, c, false, false);
    }
    int d = dt * 16 + loc;
    float bov = bo[d];
    #pragma unroll
    for (int r = 0; r < 8; ++r) {
      int lj = r + hi * 8;               // row within tile -> local j
      out[(((size_t)(b * I_ + i0 + rt)) * I_ + (j0 + lj)) * D_ + d] = c[r] + bov;
    }
  }
}

// ---------------------------------------------------------------------------
extern "C" void kernel_launch(void* const* d_in, const int* in_sizes, int n_in,
                              void* d_out, int out_size, void* d_ws, size_t ws_size,
                              hipStream_t stream) {
  const float* x    = (const float*)d_in[0];
  const int*   mask = (const int*)d_in[1];
  const float* Wl   = (const float*)d_in[2];
  const float* bl   = (const float*)d_in[3];
  const float* Wr   = (const float*)d_in[4];
  const float* br   = (const float*)d_in[5];
  const float* Wo   = (const float*)d_in[6];
  const float* bo   = (const float*)d_in[7];
  float* out = (float*)d_out;

  char* ws = (char*)d_ws;
  __bf16* wlw = (__bf16*)(ws + WLW_OFF);
  __bf16* wrw = (__bf16*)(ws + WRW_OFF);
  __bf16* wow = (__bf16*)(ws + WOW_OFF);
  float*  Lw  = (float*)(ws + LW_OFF);
  float*  Rw  = (float*)(ws + RW_OFF);

  hipLaunchKernelGGL(prep_weights_kernel, dim3(1), dim3(256), 0, stream,
                     Wl, Wr, Wo, wlw, wrw, wow);
  hipLaunchKernelGGL(proj_kernel, dim3(B_ * M_ * TI), dim3(32), 0, stream,
                     x, mask, wlw, bl, wrw, br, Lw, Rw);
  hipLaunchKernelGGL(outer_out_kernel, dim3(B_ * TI * TI), dim3(256), 0, stream,
                     Lw, Rw, wow, mask, bo, out);
}
